// ToyTransformer_36704790512182
// MI455X (gfx1250) — compile-verified, matrix operand order checked by reading, not run on hardware
//
#include <hip/hip_runtime.h>

// ---------------------------------------------------------------------------
// ToyTransformer forward for MI455X (gfx1250, wave32, WMMA).
// GEMMs: bf16 WMMA, f32 accumulate, LDS double-buffered with CDNA5 async
// paths: A tiles via GLOBAL_LOAD_ASYNC_TO_LDS_B128 (ASYNCcnt), B (weight)
// tiles via the Tensor Data Mover (TENSOR_LOAD_TO_LDS, TENSORcnt).
// ---------------------------------------------------------------------------

#define HIDDEN   1024
#define NUM_HEAD 16
#define HEAD_DIM 64
#define VOCAB    32000
#define EMBED    512
#define BATCH    4
#define SEQ      1024
#define NTOK     (BATCH * SEQ)   // 4096

typedef __bf16 bf16;
typedef __attribute__((ext_vector_type(16))) __bf16 v16bf;
typedef __attribute__((ext_vector_type(8)))  float  v8f;
typedef __attribute__((ext_vector_type(4)))  unsigned int su4;
typedef __attribute__((ext_vector_type(8)))  unsigned int su8;

union ABFrag {          // one 16x32 (A) or 32x16 (B) bf16 operand = 8 VGPRs
  v16bf v;
  uint4 u[2];
};

static __device__ inline v8f vzero8() {
  v8f c;
  #pragma unroll
  for (int i = 0; i < 8; i++) c[i] = 0.0f;
  return c;
}

static __device__ inline v8f wmma_bf16(v16bf a, v16bf b, v8f c) {
  // 8 args: (neg_a, A, neg_b, B, c_mod, C, reuse_a, reuse_b)
  return __builtin_amdgcn_wmma_f32_16x16x32_bf16(false, a, false, b,
                                                 (short)0, c, false, false);
}

static __device__ inline unsigned short f2bfu(float f) {
  bf16 h = (bf16)f;
  unsigned short s;
  __builtin_memcpy(&s, &h, 2);
  return s;
}

// --- CDNA5 async copy: global -> LDS, 16 bytes, bypasses VGPRs (ASYNCcnt) ---
static __device__ inline void async_b128(const bf16* gsrc, bf16* ldst) {
  unsigned lds = (unsigned)(unsigned long long)(void*)ldst; // low32 = LDS offset
  unsigned long long ga = (unsigned long long)gsrc;
  asm volatile("global_load_async_to_lds_b128 %0, %1, off"
               :: "v"(lds), "v"(ga) : "memory");
}
#define S_WAIT_ASYNCCNT(n) asm volatile("s_wait_asynccnt " #n ::: "memory")

// --- CDNA5 TDM: 2D tile (rows x 32 bf16), row stride K, into LDS ----------
static __device__ inline void tdm_load_tile_b(const bf16* gsrc, bf16* ldst,
                                              int K, int rows) {
  unsigned long long ga = (unsigned long long)gsrc;
  su4 g0;
  g0.x = 1u;                                        // count=1, user mode
  g0.y = (unsigned)(unsigned long long)(void*)ldst; // lds_addr
  g0.z = (unsigned)ga;                              // global_addr[31:0]
  g0.w = (unsigned)(ga >> 32) | (2u << 30);         // ga[56:32] | type=2
  su8 g1;
  g1.s0 = 0x00010000u;              // wg_mask=0, data_size=1 (2 bytes)
  g1.s1 = (32u << 16);              // tensor_dim0 = 32 (elements)
  g1.s2 = ((unsigned)rows << 16);   // tensor_dim1 = rows
  g1.s3 = (32u << 16);              // tile_dim0 = 32
  g1.s4 = (unsigned)rows;           // tile_dim1 = rows
  g1.s5 = (unsigned)K;              // tensor_dim0_stride = K (elements)
  g1.s6 = 0u;
  g1.s7 = 0u;
  asm volatile("tensor_load_to_lds %0, %1" :: "s"(g0), "s"(g1) : "memory");
}

// ---------------------------------------------------------------------------
// f32 -> bf16 conversion (weights), 4 elements / thread
// ---------------------------------------------------------------------------
__global__ void k_f32_to_bf16(const float* __restrict__ src,
                              bf16* __restrict__ dst, int n4) {
  int i = blockIdx.x * blockDim.x + threadIdx.x;
  if (i >= n4) return;
  float4 f = ((const float4*)src)[i];
  ushort4 o;
  o.x = f2bfu(f.x); o.y = f2bfu(f.y); o.z = f2bfu(f.z); o.w = f2bfu(f.w);
  *(ushort4*)(dst + (size_t)i * 4) = o;
}

// ---------------------------------------------------------------------------
// embeddings: e[tok, j] = bf16(W_word[ids[tok], j] + W_pos[tok % SEQ, j])
// ---------------------------------------------------------------------------
__global__ void k_embed(const int* __restrict__ ids,
                        const float* __restrict__ Wword,
                        const float* __restrict__ Wpos,
                        bf16* __restrict__ e) {
  int gid = blockIdx.x * blockDim.x + threadIdx.x;   // 4 elems / thread
  int tok = gid >> 7;                                // EMBED/4 = 128 chunks
  if (tok >= NTOK) return;
  int jc  = (gid & 127) << 2;
  int s   = tok & (SEQ - 1);
  int id  = ids[tok];
  float4 a = *(const float4*)(Wword + (size_t)id * EMBED + jc);
  float4 b = *(const float4*)(Wpos  + (size_t)s  * EMBED + jc);
  ushort4 o;
  o.x = f2bfu(a.x + b.x); o.y = f2bfu(a.y + b.y);
  o.z = f2bfu(a.z + b.z); o.w = f2bfu(a.w + b.w);
  *(ushort4*)(e + (size_t)tok * EMBED + jc) = o;
}

// ---------------------------------------------------------------------------
// Generic WMMA GEMM: Y[M,N] = epilogue( A[M,K] @ W[N,K]^T )
//   block tile 128x128, 8 waves, each wave 32x64 (2x4 wmma acc tiles).
//   K staged in 32-wide double-buffered LDS slabs:
//     A slab: per-lane async-to-LDS b128 (2 / lane / slab)
//     B slab: one TDM descriptor issued by wave 0
//   M,N % 128 == 0, K % 32 == 0. vt_mode stores V transposed per head.
// ---------------------------------------------------------------------------
template<bool F32OUT>
__global__ __launch_bounds__(256) void k_gemm(
    const bf16* __restrict__ A,      // [M,K]
    const bf16* __restrict__ W,      // [N,K]
    void* __restrict__ Yout,         // bf16 [M,N] / float [M,N] / vt
    const bf16* __restrict__ resid,  // [M,N] or nullptr
    int M, int N, int K, int relu, int vt_mode) {
  __shared__ bf16 lA[2][128 * 32];
  __shared__ bf16 lB[2][128 * 32];

  const int t    = threadIdx.x;
  const int lane = t & 31;
  const int w    = t >> 5;
  const int wr   = w & 3;          // wave row   (4 x 32 rows)
  const int wc   = w >> 2;         // wave col   (2 x 64 cols)
  const int bm   = blockIdx.y * 128;
  const int bn   = blockIdx.x * 128;
  const int ml   = lane & 15;      // row (A) / col (B) within 16-tile
  const int hk   = lane >> 4;      // K-half select

  v8f acc[2][4];
  #pragma unroll
  for (int i = 0; i < 2; i++)
    #pragma unroll
    for (int j = 0; j < 4; j++) acc[i][j] = vzero8();

  const int srow  = t >> 1;        // 2 threads per staged A row
  const int shalf = t & 1;         // each stages 16 bf16 = 2x b128

  auto issue_slab = [&](int k0, int buf) {
    const bf16* g = A + (size_t)(bm + srow) * K + k0 + shalf * 16;
    bf16* l = &lA[buf][srow * 32 + shalf * 16];
    async_b128(g, l);
    async_b128(g + 8, l + 8);
    if (w == 0)
      tdm_load_tile_b(W + (size_t)bn * K + k0, &lB[buf][0], K, 128);
  };

  const int nslab = K >> 5;
  issue_slab(0, 0);

  for (int i = 0; i < nslab; i++) {
    const int p = i & 1;
    const bool more = (i + 1 < nslab);
    if (more) issue_slab((i + 1) << 5, p ^ 1);

    // wait: current slab resident (next slab may remain in flight)
    if (more) S_WAIT_ASYNCCNT(2); else S_WAIT_ASYNCCNT(0);
    if (w == 0) {
      if (more) __builtin_amdgcn_s_wait_tensorcnt(1);
      else      __builtin_amdgcn_s_wait_tensorcnt(0);
    }
    __syncthreads();

    // per-wave fragment loads (ISA 16-bit A/B VGPR layout)
    ABFrag a[2], b[4];
    #pragma unroll
    for (int ii = 0; ii < 2; ii++) {
      int r = wr * 32 + ii * 16 + ml;
      a[ii].u[0] = *(const uint4*)(&lA[p][r * 32 + hk * 8]);
      a[ii].u[1] = *(const uint4*)(&lA[p][r * 32 + 16 + hk * 8]);
    }
    #pragma unroll
    for (int j = 0; j < 4; j++) {
      int n = wc * 64 + j * 16 + ml;
      b[j].u[0] = *(const uint4*)(&lB[p][n * 32 + hk * 8]);
      b[j].u[1] = *(const uint4*)(&lB[p][n * 32 + 16 + hk * 8]);
    }
    #pragma unroll
    for (int ii = 0; ii < 2; ii++)
      #pragma unroll
      for (int j = 0; j < 4; j++)
        acc[ii][j] = wmma_bf16(a[ii].v, b[j].v, acc[ii][j]);
    __syncthreads();
  }

  // ---- epilogue: D layout (m = v + 8*hk, n = ml) ----
  #pragma unroll
  for (int i = 0; i < 2; i++)
    #pragma unroll
    for (int j = 0; j < 4; j++)
      #pragma unroll
      for (int v = 0; v < 8; v++) {
        int mg = bm + wr * 32 + i * 16 + v + 8 * hk;
        int ng = bn + wc * 64 + j * 16 + ml;
        float val = acc[i][j][v];
        if (resid) val += (float)resid[(size_t)mg * N + ng];
        if (relu)  val = val > 0.0f ? val : 0.0f;
        if constexpr (F32OUT) {
          ((float*)Yout)[(size_t)mg * N + ng] = val;
        } else if (vt_mode) {
          int bi = mg >> 10, si = mg & 1023;    // token -> (batch, seq)
          int hh = ng >> 6,  dd = ng & 63;      // col   -> (head, dim)
          ((bf16*)Yout)[(((size_t)bi * NUM_HEAD + hh) * HEAD_DIM + dd) * SEQ + si] =
              (bf16)val;
        } else {
          ((bf16*)Yout)[(size_t)mg * N + ng] = (bf16)val;
        }
      }
}

// ---------------------------------------------------------------------------
// Flash attention: one wave per (batch, head, 16-row Q tile).
//   scores = Q K^T / sqrt(HIDDEN), causal, online softmax, O = P V.
//   Only the single diagonal 32-key block is masked; all earlier blocks
//   take a mask-free fast path.
//   q,k: [NTOK, HIDDEN] bf16;  vt: [B*H, HEAD_DIM, SEQ] bf16 (pre-transposed)
// ---------------------------------------------------------------------------
__global__ __launch_bounds__(32) void k_attn(const bf16* __restrict__ q,
                                             const bf16* __restrict__ k,
                                             const bf16* __restrict__ vt,
                                             bf16* __restrict__ out) {
  const int lane = threadIdx.x & 31;
  const int ml = lane & 15, hk = lane >> 4;
  const int qb = blockIdx.x * 16;
  const int h  = blockIdx.y;
  const int b  = blockIdx.z;
  __shared__ bf16 lp[16 * 32];   // P patch (D-layout -> A-layout round trip)

  // Q operand (16 x 64): two K-steps of 32
  ABFrag aq0, aq1;
  {
    const bf16* qrow = q + ((size_t)(b * SEQ + qb + ml)) * HIDDEN + h * HEAD_DIM;
    aq0.u[0] = *(const uint4*)(qrow + hk * 8);
    aq0.u[1] = *(const uint4*)(qrow + 16 + hk * 8);
    aq1.u[0] = *(const uint4*)(qrow + 32 + hk * 8);
    aq1.u[1] = *(const uint4*)(qrow + 48 + hk * 8);
  }

  v8f o[4];
  #pragma unroll
  for (int j = 0; j < 4; j++) o[j] = vzero8();
  float rm[8], rl[8];
  #pragma unroll
  for (int v = 0; v < 8; v++) { rm[v] = -1e30f; rl[v] = 0.0f; }

  const bf16* kbase = k  + ((size_t)b * SEQ) * HIDDEN + h * HEAD_DIM;
  const bf16* vbase = vt + ((size_t)(b * NUM_HEAD + h)) * HEAD_DIM * SEQ;

  auto process = [&](int j0, bool domask) {
    // S tiles: 16 q x 32 keys as two 16x16 wmma results
    v8f s[2];
    #pragma unroll
    for (int tt = 0; tt < 2; tt++) {
      const bf16* krow = kbase + (size_t)(j0 + tt * 16 + ml) * HIDDEN;
      ABFrag b0, b1;
      b0.u[0] = *(const uint4*)(krow + hk * 8);
      b0.u[1] = *(const uint4*)(krow + 16 + hk * 8);
      b1.u[0] = *(const uint4*)(krow + 32 + hk * 8);
      b1.u[1] = *(const uint4*)(krow + 48 + hk * 8);
      v8f c = vzero8();
      c = wmma_bf16(aq0.v, b0.v, c);
      c = wmma_bf16(aq1.v, b1.v, c);
      s[tt] = c;
    }

    // scale by 1/sqrt(HIDDEN)=1/32; causal mask only on diagonal block
    #pragma unroll
    for (int tt = 0; tt < 2; tt++)
      #pragma unroll
      for (int v = 0; v < 8; v++) {
        float x = s[tt][v] * 0.03125f;
        if (domask) {
          int kn = j0 + tt * 16 + ml;
          int qm = qb + v + 8 * hk;
          x = (kn > qm) ? -1e30f : x;
        }
        s[tt][v] = x;
      }

    // online softmax: 16-lane row reductions (rows live in half-waves)
    float alpha[8];
    #pragma unroll
    for (int v = 0; v < 8; v++) {
      float m = fmaxf(s[0][v], s[1][v]);
      #pragma unroll
      for (int off = 8; off >= 1; off >>= 1)
        m = fmaxf(m, __shfl_xor(m, off, 32));
      float mn = fmaxf(rm[v], m);
      alpha[v] = __expf(rm[v] - mn);
      rm[v] = mn;
    }
    #pragma unroll
    for (int v = 0; v < 8; v++) {
      float p0 = __expf(s[0][v] - rm[v]);
      float p1 = __expf(s[1][v] - rm[v]);
      s[0][v] = p0; s[1][v] = p1;
      float rs = p0 + p1;
      #pragma unroll
      for (int off = 8; off >= 1; off >>= 1)
        rs += __shfl_xor(rs, off, 32);
      rl[v] = rl[v] * alpha[v] + rs;
    }
    #pragma unroll
    for (int j = 0; j < 4; j++)
      #pragma unroll
      for (int v = 0; v < 8; v++) o[j][v] *= alpha[v];

    // P: D-layout regs -> LDS -> A-layout operand (16 x 32)
    #pragma unroll
    for (int tt = 0; tt < 2; tt++)
      #pragma unroll
      for (int v = 0; v < 8; v++)
        lp[(v + 8 * hk) * 32 + tt * 16 + ml] = (bf16)s[tt][v];
    __syncthreads();
    ABFrag ap;
    ap.u[0] = *(const uint4*)(lp + ml * 32 + hk * 8);
    ap.u[1] = *(const uint4*)(lp + ml * 32 + 16 + hk * 8);
    __syncthreads();

    // O += P @ V : vt rows are K-contiguous -> b128 loads
    #pragma unroll
    for (int j = 0; j < 4; j++) {
      const bf16* vrow = vbase + (size_t)(j * 16 + ml) * SEQ + j0;
      ABFrag bv;
      bv.u[0] = *(const uint4*)(vrow + hk * 8);
      bv.u[1] = *(const uint4*)(vrow + 16 + hk * 8);
      o[j] = wmma_bf16(ap.v, bv.v, o[j]);
    }
  };

  const int nfull = qb >> 5;                 // fully-unmasked 32-key blocks
  for (int jb = 0; jb < nfull; jb++) process(jb * 32, false);
  process(nfull * 32, true);                 // single diagonal block

  // finalize: O /= l, store to attn_out[tok, h*64 + d]
  #pragma unroll
  for (int v = 0; v < 8; v++) rl[v] = 1.0f / rl[v];
  #pragma unroll
  for (int j = 0; j < 4; j++)
    #pragma unroll
    for (int v = 0; v < 8; v++) {
      int mg = qb + v + 8 * hk;
      out[(size_t)(b * SEQ + mg) * HIDDEN + h * HEAD_DIM + j * 16 + ml] =
          (bf16)(o[j][v] * rl[v]);
    }
}

// ---------------------------------------------------------------------------
// host-side orchestration
// ---------------------------------------------------------------------------
extern "C" void kernel_launch(void* const* d_in, const int* in_sizes, int n_in,
                              void* d_out, int out_size, void* d_ws, size_t ws_size,
                              hipStream_t stream) {
  const int*   ids   = (const int*)  d_in[0];
  const float* Wword = (const float*)d_in[1];
  const float* Wpos  = (const float*)d_in[2];
  const float* Wq    = (const float*)d_in[3];
  const float* Wk    = (const float*)d_in[4];
  const float* Wv    = (const float*)d_in[5];
  const float* Wo    = (const float*)d_in[6];
  const float* W1    = (const float*)d_in[7];
  const float* W2    = (const float*)d_in[8];
  const float* Weh   = (const float*)d_in[9];
  const float* Whe   = (const float*)d_in[10];
  const float* Wlm   = (const float*)d_in[11];
  (void)in_sizes; (void)n_in; (void)out_size; (void)ws_size;

  char* wp = (char*)d_ws;
  auto alloc = [&](size_t bytes) -> void* {
    void* p = wp;
    wp += (bytes + 255) & ~(size_t)255;
    return p;
  };

  // bf16 weight copies
  bf16* wq_b  = (bf16*)alloc(2ull * HIDDEN * HIDDEN);
  bf16* wk_b  = (bf16*)alloc(2ull * HIDDEN * HIDDEN);
  bf16* wv_b  = (bf16*)alloc(2ull * HIDDEN * HIDDEN);
  bf16* wo_b  = (bf16*)alloc(2ull * HIDDEN * HIDDEN);
  bf16* w1_b  = (bf16*)alloc(2ull * 4 * HIDDEN * HIDDEN);
  bf16* w2_b  = (bf16*)alloc(2ull * HIDDEN * 4 * HIDDEN);
  bf16* weh_b = (bf16*)alloc(2ull * HIDDEN * EMBED);
  bf16* whe_b = (bf16*)alloc(2ull * EMBED * HIDDEN);
  bf16* wlm_b = (bf16*)alloc(2ull * VOCAB * EMBED);
  // bf16 activations
  bf16* e_b   = (bf16*)alloc(2ull * NTOK * EMBED);
  bf16* h0_b  = (bf16*)alloc(2ull * NTOK * HIDDEN);
  bf16* q_b   = (bf16*)alloc(2ull * NTOK * HIDDEN);
  bf16* k_b   = (bf16*)alloc(2ull * NTOK * HIDDEN);
  bf16* vt_b  = (bf16*)alloc(2ull * NTOK * HIDDEN);   // [B*H, 64, SEQ]
  bf16* ao_b  = (bf16*)alloc(2ull * NTOK * HIDDEN);
  bf16* h1_b  = (bf16*)alloc(2ull * NTOK * HIDDEN);
  bf16* x_b   = (bf16*)alloc(2ull * NTOK * 4 * HIDDEN);
  bf16* h2_b  = (bf16*)alloc(2ull * NTOK * HIDDEN);
  bf16* he_b  = (bf16*)alloc(2ull * NTOK * EMBED);

  auto conv = [&](const float* src, bf16* dst, size_t n) {
    int n4 = (int)(n / 4);
    k_f32_to_bf16<<<(n4 + 255) / 256, 256, 0, stream>>>(src, dst, n4);
  };
  conv(Wq,  wq_b,  (size_t)HIDDEN * HIDDEN);
  conv(Wk,  wk_b,  (size_t)HIDDEN * HIDDEN);
  conv(Wv,  wv_b,  (size_t)HIDDEN * HIDDEN);
  conv(Wo,  wo_b,  (size_t)HIDDEN * HIDDEN);
  conv(W1,  w1_b,  4ull * HIDDEN * HIDDEN);
  conv(W2,  w2_b,  4ull * HIDDEN * HIDDEN);
  conv(Weh, weh_b, (size_t)HIDDEN * EMBED);
  conv(Whe, whe_b, (size_t)EMBED * HIDDEN);
  conv(Wlm, wlm_b, (size_t)VOCAB * EMBED);

  // embeddings
  {
    int nthr = NTOK * EMBED / 4;
    k_embed<<<(nthr + 255) / 256, 256, 0, stream>>>(ids, Wword, Wpos, e_b);
  }

  auto gemm = [&](const bf16* A, const bf16* W, void* Y, const bf16* R,
                  int M, int N, int K, int relu, int vt, bool f32out) {
    dim3 g(N / 128, M / 128);
    if (f32out)
      k_gemm<true><<<g, 256, 0, stream>>>(A, W, Y, R, M, N, K, relu, vt);
    else
      k_gemm<false><<<g, 256, 0, stream>>>(A, W, Y, R, M, N, K, relu, vt);
  };

  // h0 = e @ Weh^T
  gemm(e_b, weh_b, h0_b, nullptr, NTOK, HIDDEN, EMBED, 0, 0, false);
  // q,k = h0 @ Wq^T / Wk^T ; v stored transposed per head
  gemm(h0_b, wq_b, q_b,  nullptr, NTOK, HIDDEN, HIDDEN, 0, 0, false);
  gemm(h0_b, wk_b, k_b,  nullptr, NTOK, HIDDEN, HIDDEN, 0, 0, false);
  gemm(h0_b, wv_b, vt_b, nullptr, NTOK, HIDDEN, HIDDEN, 0, 1, false);
  // attention
  k_attn<<<dim3(SEQ / 16, NUM_HEAD, BATCH), 32, 0, stream>>>(q_b, k_b, vt_b, ao_b);
  // h1 = h0 + ao @ Wo^T
  gemm(ao_b, wo_b, h1_b, h0_b, NTOK, HIDDEN, HIDDEN, 0, 0, false);
  // x = relu(h1 @ W1^T)
  gemm(h1_b, w1_b, x_b, nullptr, NTOK, 4 * HIDDEN, HIDDEN, 1, 0, false);
  // h2 = h1 + x @ W2^T
  gemm(x_b, w2_b, h2_b, h1_b, NTOK, HIDDEN, 4 * HIDDEN, 0, 0, false);
  // he = h2 @ Whe^T
  gemm(h2_b, whe_b, he_b, nullptr, NTOK, EMBED, HIDDEN, 0, 0, false);
  // logits = he @ Wlm^T  (f32 straight to d_out)
  gemm(he_b, wlm_b, d_out, nullptr, NTOK, VOCAB, EMBED, 0, 0, true);
}
